// IternetMult_63848983823190
// MI455X (gfx1250) — matrix operand order repaired and checked by reading, not compiled, not run on hardware
//
#include <hip/hip_runtime.h>
#include <math.h>

typedef __attribute__((ext_vector_type(16))) _Float16 v16h;
typedef __attribute__((ext_vector_type(8)))  _Float16 v8h;
typedef __attribute__((ext_vector_type(8)))  float    v8f;

#define WIDTH     1024
#define HID_ROWS  1027      // 3 + 1024
#define MBLK      64        // rows per workgroup
#define THREADS   512       // 16 waves (wave32)
#define NWAVES    16
#define MTILES    (MBLK/16)     // 4
#define NT        4             // n-tiles per wave (64 cols / wave, 16 waves * 64 = 1024)
// Padded LDS row stride: 2064 B = 516 dwords == 4 (mod 64 banks) -> conflict-free
// ds_load_b128 A-tile reads (16 lanes x 4 banks tile all 64 banks exactly once).
#define STRIDE    (WIDTH + 8)

// ---------------- prep kernel 1: column norms + normalized w_in rows ----------------
__global__ void prep_norm(const float* __restrict__ w, float* __restrict__ w012,
                          float* __restrict__ invn) {
    int n = blockIdx.x * blockDim.x + threadIdx.x;
    if (n >= WIDTH) return;
    float s = 0.f;
    for (int k = 0; k < HID_ROWS; ++k) { float v = w[k * WIDTH + n]; s += v * v; }
    float inv = 1.0f / sqrtf(s);
    invn[n] = inv;
    w012[0 * WIDTH + n] = w[0 * WIDTH + n] * inv;
    w012[1 * WIDTH + n] = w[1 * WIDTH + n] * inv;
    w012[2 * WIDTH + n] = w[2 * WIDTH + n] * inv;
}

// ---------------- prep kernel 2: transpose + normalize + cast to f16 ----------------
// wT[n*WIDTH + k] = (f16)( hidden_w[(3+k)*WIDTH + n] * invn[n] )
__global__ void prep_transpose(const float* __restrict__ w, const float* __restrict__ invn,
                               _Float16* __restrict__ wT) {
    __shared__ float tile[32][33];
    int k0 = blockIdx.x * 32, n0 = blockIdx.y * 32;
    int tx = threadIdx.x, ty = threadIdx.y;
    #pragma unroll
    for (int i = 0; i < 4; ++i) {
        int k = k0 + ty * 4 + i;
        tile[ty * 4 + i][tx] = w[(size_t)(3 + k) * WIDTH + (n0 + tx)];
    }
    __syncthreads();
    #pragma unroll
    for (int i = 0; i < 4; ++i) {
        int n = n0 + ty * 4 + i;
        wT[(size_t)n * WIDTH + (k0 + tx)] = (_Float16)(tile[tx][ty * 4 + i] * invn[n]);
    }
}

// ---------------- main kernel: all iterations for a 64-row block ----------------
__global__ __launch_bounds__(THREADS) void iternet_main(
    const float* __restrict__ inp, const float* __restrict__ lastw,
    const int* __restrict__ itersPtr,
    const _Float16* __restrict__ wT, const float* __restrict__ w012,
    float* __restrict__ out)
{
    __shared__ _Float16 hid[MBLK * STRIDE];  // 64*1032*2 ~= 129 KB (CDNA5: up to 320 KB/WG)
    __shared__ float2   xy[MBLK];

    const int tid     = threadIdx.x;
    const int lane    = tid & 31;
    const int wave    = tid >> 5;        // 0..15
    const int lane_lo = lane & 15;
    const int lane_hi = lane >> 4;       // 0 or 1
    const int m0      = blockIdx.x * MBLK;
    const int nbase   = wave * 64;       // this wave's column strip

    if (tid < MBLK) xy[tid] = ((const float2*)inp)[m0 + tid];
    __syncthreads();

    const int iters = *itersPtr;

    // fixed-term coefficients for this wave's columns (same col for every acc row)
    float c0[NT], c1[NT], c2[NT];
    #pragma unroll
    for (int nt = 0; nt < NT; ++nt) {
        int col = nbase + nt * 16 + lane_lo;
        c0[nt] = w012[0 * WIDTH + col];
        c1[nt] = w012[1 * WIDTH + col];
        c2[nt] = w012[2 * WIDTH + col];
    }

    v8f acc[MTILES][NT];

    // relu(fixed + acc) -> f16 -> LDS.  C/D layout: VGPR v, lanes0-15: M=v N=lane;
    // lanes16-31: M=v+8 N=lane-16.
    auto storePhase = [&]() {
        #pragma unroll
        for (int mt = 0; mt < MTILES; ++mt) {
            #pragma unroll
            for (int v = 0; v < 8; ++v) {
                int row = mt * 16 + v + lane_hi * 8;
                float2 p = xy[row];
                #pragma unroll
                for (int nt = 0; nt < NT; ++nt) {
                    int col = nbase + nt * 16 + lane_lo;
                    float val = acc[mt][nt][v] + c0[nt] + p.x * c1[nt] + p.y * c2[nt];
                    val = fmaxf(val, 0.0f);
                    hid[row * STRIDE + col] = (_Float16)val;
                }
            }
        }
    };

    auto zeroAcc = [&]() {
        #pragma unroll
        for (int mt = 0; mt < MTILES; ++mt)
            #pragma unroll
            for (int nt = 0; nt < NT; ++nt)
                acc[mt][nt] = 0.0f;
    };

    // ---- iteration 0: hidden = relu(fixed) ----
    zeroAcc();
    storePhase();

    // ---- iterations 1..iters-1: hidden = relu(fixed + hidden @ Whid) ----
    for (int it = 1; it < iters; ++it) {
        __syncthreads();          // stores visible before reads
        zeroAcc();
        for (int k0 = 0; k0 < WIDTH; k0 += 32) {
            // B tiles (32x16 KxN): lane = column nbase+nt*16+lane_lo,
            // lanes0-15 hold K=k0..k0+15, lanes16-31 K=k0+16..k0+31 (contiguous in wT).
            v16h b[NT];
            #pragma unroll
            for (int nt = 0; nt < NT; ++nt) {
                const _Float16* bp = wT + (size_t)(nbase + nt * 16 + lane_lo) * WIDTH
                                        + k0 + lane_hi * 16;
                b[nt] = *(const v16h*)bp;
            }
            // A tiles (16x32 MxK), one private register buffer per m-tile so the
            // refill next k-step lands >= 12 WMMAs after the last read (no hazard NOPs).
            union { v16h v; v8h h[2]; } A[MTILES];
            #pragma unroll
            for (int mt = 0; mt < MTILES; ++mt) {
                const _Float16* ap = &hid[(mt * 16 + lane_lo) * STRIDE + k0 + lane_hi * 16];
                A[mt].h[0] = *(const v8h*)ap;
                A[mt].h[1] = *(const v8h*)(ap + 8);
            }
            #pragma unroll
            for (int mt = 0; mt < MTILES; ++mt) {
                #pragma unroll
                for (int nt = 0; nt < NT; ++nt) {
                    acc[mt][nt] = __builtin_amdgcn_wmma_f32_16x16x32_f16(
                        false, A[mt].v, false, b[nt], (short)0, acc[mt][nt], false, false);
                }
            }
        }
        __syncthreads();          // all reads done before overwriting hidden
        storePhase();
    }

    // ---- out = hidden @ last ----
    __syncthreads();
    #pragma unroll
    for (int rr = 0; rr < MBLK / NWAVES; ++rr) {
        int r = wave * (MBLK / NWAVES) + rr;
        float s = 0.0f;
        #pragma unroll 8
        for (int j = 0; j < WIDTH / 32; ++j) {
            int c = lane + j * 32;
            s += (float)hid[r * STRIDE + c] * lastw[c];
        }
        #pragma unroll
        for (int off = 16; off > 0; off >>= 1)
            s += __shfl_down(s, off, 32);
        if (lane == 0) out[m0 + r] = s;
    }
}

extern "C" void kernel_launch(void* const* d_in, const int* in_sizes, int n_in,
                              void* d_out, int out_size, void* d_ws, size_t ws_size,
                              hipStream_t stream) {
    const float* inp   = (const float*)d_in[0];   // (8192, 2)
    const float* hw    = (const float*)d_in[1];   // (1027, 1024)
    const float* lastw = (const float*)d_in[2];   // (1024, 1)
    const int*   iters = (const int*)d_in[3];     // scalar
    float* out = (float*)d_out;

    char* ws = (char*)d_ws;
    _Float16* wT   = (_Float16*)ws;                                   // 2 MB
    float*    w012 = (float*)(ws + (size_t)WIDTH * WIDTH * sizeof(_Float16));
    float*    invn = w012 + 3 * WIDTH;

    prep_norm<<<WIDTH / 256, 256, 0, stream>>>(hw, w012, invn);
    prep_transpose<<<dim3(WIDTH / 32, WIDTH / 32), dim3(32, 8), 0, stream>>>(hw, invn, wT);

    int B = in_sizes[0] / 2;   // 8192 rows
    iternet_main<<<B / MBLK, THREADS, 0, stream>>>(inp, lastw, iters, wT, w012, out);
}
